// AttentionLayer_39161511804956
// MI455X (gfx1250) — matrix operand order repaired and checked by reading, not compiled, not run on hardware
//
#include <hip/hip_runtime.h>
#include <hip/hip_bf16.h>
#include <math.h>

// Problem constants
#define B_  8
#define V_  1024
#define D_  512
#define H_  8
#define HD_ 64

typedef __attribute__((ext_vector_type(16))) __bf16 v16bf;
typedef __attribute__((ext_vector_type(8)))  float  v8f;

union ABfrag {
  v16bf v;
  unsigned short u[16];
  uint4 q[2];
};

__device__ __forceinline__ unsigned short f2bf(float f) {
  unsigned int u = __float_as_uint(f);
  u += 0x7FFFu + ((u >> 16) & 1u);   // round-to-nearest-even
  return (unsigned short)(u >> 16);
}
__device__ __forceinline__ float bf2f(unsigned short h) {
  return __uint_as_float(((unsigned int)h) << 16);
}
__device__ __forceinline__ v8f wmma_bf16(v16bf a, v16bf b, v8f c) {
  return __builtin_amdgcn_wmma_f32_16x16x32_bf16(false, a, false, b, (short)0, c,
                                                 false, false);
}

// ---- workspace layout (u16 element offsets) -------------------------------
#define XB_OFF     0u         // x bf16                [8192][512]
#define WQKVT_OFF  4194304u   // w_qkv^T bf16          [1536][512]
#define WOUTT_OFF  4980736u   // w_out^T bf16          [512][512]
#define QH_OFF     5242880u   // Q  bf16  [b][h][v][hd]
#define KH_OFF     9437184u   // K  bf16  [b][h][v][hd]
#define VT_OFF     13631488u  // V^T bf16 [b][h][hd][v]
#define HEADS_OFF  17825792u  // out_heads bf16 [b][v][h*64+hd] = [8192][512]
// total 22,020,096 u16 = ~44 MB

// ---------------------------------------------------------------------------
// Kernel 1: convert x -> bf16, transpose+convert w_qkv and w_out -> bf16
// ---------------------------------------------------------------------------
__global__ void k_convert(const float* __restrict__ x,
                          const float* __restrict__ w_qkv,
                          const float* __restrict__ w_out,
                          unsigned short* __restrict__ W) {
  const unsigned int NX = 4194304u, NQ = 786432u, NW = 262144u;
  const unsigned int stride = gridDim.x * blockDim.x;
  for (unsigned int i = blockIdx.x * blockDim.x + threadIdx.x;
       i < NX + NQ + NW; i += stride) {
    if (i < NX) {
      W[XB_OFF + i] = f2bf(x[i]);
    } else if (i < NX + NQ) {
      unsigned int j = i - NX;
      unsigned int r = j / 1536u, c = j % 1536u;
      W[WQKVT_OFF + c * 512u + r] = f2bf(w_qkv[j]);
    } else {
      unsigned int j = i - NX - NQ;
      unsigned int r = j >> 9, c = j & 511u;
      W[WOUTT_OFF + c * 512u + r] = f2bf(w_out[j]);
    }
  }
}

// ---------------------------------------------------------------------------
// Kernel 2: qkv = x @ w_qkv  (bf16 WMMA, f32 acc), scatter into Q/K/V^T
// Register-blocked: each wave computes a 32x64 output tile
// (2 A frags + 4 B frags -> 8 WMMAs per K-step of 32).
// ---------------------------------------------------------------------------
__global__ void __launch_bounds__(256) k_qkv(unsigned short* __restrict__ W) {
  const unsigned short* xb = W + XB_OFF;
  const unsigned short* wq = W + WQKVT_OFF;
  const unsigned int lane = threadIdx.x & 31u;
  const unsigned int wave = threadIdx.x >> 5;
  const unsigned int tile = blockIdx.x * 8u + wave;      // < 256*24 = 6144
  const unsigned int mt = tile / 24u, nt = tile % 24u;
  const unsigned int row0 = mt * 32u, col0 = nt * 64u;
  const unsigned int l15 = lane & 15u;
  const unsigned int halfsel = lane >> 4;
  const unsigned int kA = halfsel * 8u;
  const unsigned int kB = halfsel * 8u;

  v8f acc[2][4];
#pragma unroll
  for (unsigned int mi = 0; mi < 2u; ++mi)
#pragma unroll
    for (unsigned int t = 0; t < 4u; ++t)
      acc[mi][t] = (v8f){0.f, 0.f, 0.f, 0.f, 0.f, 0.f, 0.f, 0.f};

  const unsigned short* apr0 = xb + (row0 + l15) * 512u + kA;
  const unsigned short* apr1 = apr0 + 16u * 512u;
  const unsigned short* bpr  = wq + (col0 + l15) * 512u + kB;

  for (unsigned int k0 = 0; k0 < 512u; k0 += 32u) {
    ABfrag a0, a1;
    a0.q[0] = *(const uint4*)(apr0 + k0);
    a0.q[1] = *(const uint4*)(apr0 + k0 + 16);
    a1.q[0] = *(const uint4*)(apr1 + k0);
    a1.q[1] = *(const uint4*)(apr1 + k0 + 16);
#pragma unroll
    for (unsigned int t = 0; t < 4u; ++t) {
      ABfrag b;
      const unsigned short* bp = bpr + t * (16u * 512u) + k0;
      b.q[0] = *(const uint4*)(bp);
      b.q[1] = *(const uint4*)(bp + 16);
      acc[0][t] = wmma_bf16(a0.v, b.v, acc[0][t]);
      acc[1][t] = wmma_bf16(a1.v, b.v, acc[1][t]);
    }
  }

  // scatter: cols [0,512)->Q, [512,1024)->K, [1024,1536)->V (transposed).
  // 64-col supertile is 64-aligned => seg and h are wave-uniform.
  const unsigned int seg = col0 >> 9;
  const unsigned int h   = (col0 & 511u) >> 6;
#pragma unroll
  for (unsigned int mi = 0; mi < 2u; ++mi) {
#pragma unroll
    for (unsigned int t = 0; t < 4u; ++t) {
      const unsigned int hd = t * 16u + l15;
#pragma unroll
      for (unsigned int r = 0; r < 8u; ++r) {
        const unsigned int M = mi * 16u + r + (halfsel << 3);
        const unsigned int grow = row0 + M;
        const unsigned int bb = grow >> 10, v = grow & 1023u;
        const unsigned short val = f2bf(acc[mi][t][r]);
        if (seg == 0u)
          W[QH_OFF + ((bb * 8u + h) * 1024u + v) * 64u + hd] = val;
        else if (seg == 1u)
          W[KH_OFF + ((bb * 8u + h) * 1024u + v) * 64u + hd] = val;
        else
          W[VT_OFF + ((bb * 8u + h) * 64u + hd) * 1024u + v] = val;
      }
    }
  }
}

// ---------------------------------------------------------------------------
// Kernel 3: fused scores -> +bias -> softmax -> attn write -> attn @ V
// One wave per (h, 16-query block); loops over b so hop_matrix is read once.
// LDS: scores f32 [16][1028] + bias bf16 [16][1024] + rpe + row-stat.
// ---------------------------------------------------------------------------
#define SC_STRIDE 1028u
#define LDS_SC_BYTES   (16u * SC_STRIDE * 4u)            // 65792
#define LDS_BIAS_BYTES (16u * 1024u * 2u)                // 32768
#define LDS_RPE_BYTES  64u
#define LDS_STAT_BYTES 64u
#define LDS_TOTAL (LDS_SC_BYTES + LDS_BIAS_BYTES + LDS_RPE_BYTES + LDS_STAT_BYTES)

__global__ void __launch_bounds__(32) k_attn(const float* __restrict__ rpe,
                                             const int* __restrict__ hop,
                                             unsigned short* __restrict__ W,
                                             float* __restrict__ out) {
  extern __shared__ char smem_raw[];
  float*          s_sc   = (float*)smem_raw;                        // 16 x 1028
  unsigned short* s_bias = (unsigned short*)(smem_raw + LDS_SC_BYTES);
  float*          s_rpe  = (float*)(smem_raw + LDS_SC_BYTES + LDS_BIAS_BYTES);
  float*          s_stat = (float*)(smem_raw + LDS_SC_BYTES + LDS_BIAS_BYTES +
                                    LDS_RPE_BYTES);

  const unsigned int lane = threadIdx.x;
  const unsigned int h  = blockIdx.x >> 6;
  const unsigned int vb = blockIdx.x & 63u;
  const unsigned int v0 = vb * 16u;

  if (lane < 9u) s_rpe[lane] = rpe[h * 9u + lane];
  __syncthreads();

  // build bias tile once (hop_matrix is batch-independent)
  for (unsigned int i = lane; i < 16u * 1024u; i += 32u) {
    const unsigned int r = i >> 10, c = i & 1023u;
    int hp = hop[(v0 + r) * 1024u + c];
    hp = hp < 0 ? 0 : (hp > 8 ? 8 : hp);
    s_bias[i] = f2bf(s_rpe[hp]);
  }
  __syncthreads();

  const float scale = 0.125f;   // HD^-0.5
  const unsigned int rowA    = lane & 15u;
  const unsigned int halfsel = lane >> 4;
  float* attn_all = out + 4194304u;   // attn region of d_out

  for (unsigned int b = 0; b < 8u; ++b) {
    const unsigned short* qb  = W + QH_OFF + ((b * 8u + h) * 1024u + v0) * 64u;
    const unsigned short* kb_ = W + KH_OFF + ((b * 8u + h) * 1024u) * 64u;
    const unsigned short* vt  = W + VT_OFF + ((b * 8u + h) * 64u) * 1024u;

    // Q fragments: K-steps hd[0,32) and hd[32,64)
    ABfrag a0, a1;
    {
      const unsigned short* qp = qb + rowA * 64u + halfsel * 8u;
      a0.q[0] = *(const uint4*)(qp);
      a0.q[1] = *(const uint4*)(qp + 16);
      a1.q[0] = *(const uint4*)(qp + 32);
      a1.q[1] = *(const uint4*)(qp + 48);
    }

    // ---- scores sweep: S = Q K^T, +bias, *scale, into LDS ----
    for (unsigned int w0 = 0; w0 < 1024u; w0 += 16u) {
      ABfrag b0, b1;
      const unsigned short* kp =
          kb_ + (w0 + (lane & 15u)) * 64u + halfsel * 8u;
      b0.q[0] = *(const uint4*)(kp);
      b0.q[1] = *(const uint4*)(kp + 16);
      b1.q[0] = *(const uint4*)(kp + 32);
      b1.q[1] = *(const uint4*)(kp + 48);
      v8f c = {0.f, 0.f, 0.f, 0.f, 0.f, 0.f, 0.f, 0.f};
      c = wmma_bf16(a0.v, b0.v, c);
      c = wmma_bf16(a1.v, b1.v, c);
      const unsigned int col = w0 + (lane & 15u);
#pragma unroll
      for (unsigned int r = 0; r < 8u; ++r) {
        const unsigned int M = r + (halfsel << 3);
        const float bias = bf2f(s_bias[M * 1024u + col]);
        s_sc[M * SC_STRIDE + col] = (c[r] + bias) * scale;
      }
    }
    __syncthreads();

    // ---- row max (lane pair (r, r+16) splits 1024 cols) ----
    float m = -3.0e38f;
    {
      const float* pr = s_sc + rowA * SC_STRIDE + halfsel * 512u;
      for (unsigned int c4 = 0; c4 < 512u; c4 += 4u) {
        const float4 t = *(const float4*)(pr + c4);
        m = fmaxf(m, fmaxf(fmaxf(t.x, t.y), fmaxf(t.z, t.w)));
      }
    }
    m = fmaxf(m, __shfl_xor(m, 16, 32));

    // ---- exp & row sum (overwrite LDS with e) ----
    float sum = 0.f;
    {
      float* pr = s_sc + rowA * SC_STRIDE + halfsel * 512u;
      for (unsigned int c4 = 0; c4 < 512u; c4 += 4u) {
        float4 t = *(float4*)(pr + c4);
        t.x = __expf(t.x - m);
        t.y = __expf(t.y - m);
        t.z = __expf(t.z - m);
        t.w = __expf(t.w - m);
        sum += t.x + t.y + t.z + t.w;
        *(float4*)(pr + c4) = t;
      }
    }
    sum += __shfl_xor(sum, 16, 32);
    const float inv = 1.f / sum;
    if (halfsel == 0u) s_stat[rowA] = inv;
    __syncthreads();

    // ---- coalesced attn write: 32 lanes cooperate per 4KB row ----
    {
      const size_t bh_base = ((size_t)((b * 8u + h) * 1024u + v0)) * 1024u;
#pragma unroll 2
      for (unsigned int r = 0; r < 16u; ++r) {
        const float rinv = s_stat[r];
        float* gp = attn_all + bh_base + (size_t)r * 1024u;
        const float* lp = s_sc + r * SC_STRIDE;
        for (unsigned int c = lane * 4u; c < 1024u; c += 128u) {
          float4 t = *(const float4*)(lp + c);
          t.x *= rinv; t.y *= rinv; t.z *= rinv; t.w *= rinv;
          *(float4*)(gp + c) = t;
        }
      }
    }

    // ---- AV sweep: attn(bf16) @ V via WMMA ----
    v8f acc[4];
#pragma unroll
    for (unsigned int t = 0; t < 4u; ++t)
      acc[t] = (v8f){0.f, 0.f, 0.f, 0.f, 0.f, 0.f, 0.f, 0.f};

    for (unsigned int kb0 = 0; kb0 < 1024u; kb0 += 32u) {
      const unsigned int k0 = kb0 + halfsel * 8u;
      const float* sp = s_sc + rowA * SC_STRIDE + k0;
      float4 x0 = *(const float4*)(sp);
      float4 x1 = *(const float4*)(sp + 4);
      float4 y0 = *(const float4*)(sp + 16);
      float4 y1 = *(const float4*)(sp + 20);

      ABfrag af;
      af.u[0]  = f2bf(x0.x * inv); af.u[1]  = f2bf(x0.y * inv);
      af.u[2]  = f2bf(x0.z * inv); af.u[3]  = f2bf(x0.w * inv);
      af.u[4]  = f2bf(x1.x * inv); af.u[5]  = f2bf(x1.y * inv);
      af.u[6]  = f2bf(x1.z * inv); af.u[7]  = f2bf(x1.w * inv);
      af.u[8]  = f2bf(y0.x * inv); af.u[9]  = f2bf(y0.y * inv);
      af.u[10] = f2bf(y0.z * inv); af.u[11] = f2bf(y0.w * inv);
      af.u[12] = f2bf(y1.x * inv); af.u[13] = f2bf(y1.y * inv);
      af.u[14] = f2bf(y1.z * inv); af.u[15] = f2bf(y1.w * inv);

#pragma unroll
      for (unsigned int t = 0; t < 4u; ++t) {
        ABfrag bv;
        const unsigned short* vp =
            vt + (t * 16u + (lane & 15u)) * 1024u + kb0 + halfsel * 8u;
        bv.q[0] = *(const uint4*)(vp);
        bv.q[1] = *(const uint4*)(vp + 16);
        acc[t] = wmma_bf16(af.v, bv.v, acc[t]);
      }
    }

    // ---- write out_heads (bf16) [b][v][h*64+hd] ----
#pragma unroll
    for (unsigned int t = 0; t < 4u; ++t) {
#pragma unroll
      for (unsigned int r = 0; r < 8u; ++r) {
        const unsigned int M = r + (halfsel << 3);
        W[HEADS_OFF + (b * 1024u + v0 + M) * 512u + h * 64u + t * 16u +
          (lane & 15u)] = f2bf(acc[t][r]);
      }
    }
    __syncthreads();
  }
}

// ---------------------------------------------------------------------------
// Kernel 4: out = heads @ w_out + b_out  (bf16 WMMA, f32 out)
// Register-blocked 32x64 per wave.
// ---------------------------------------------------------------------------
__global__ void __launch_bounds__(256) k_out(const unsigned short* __restrict__ W,
                                             const float* __restrict__ b_out,
                                             float* __restrict__ out) {
  const unsigned short* Aa = W + HEADS_OFF;
  const unsigned short* Bb = W + WOUTT_OFF;
  const unsigned int lane = threadIdx.x & 31u;
  const unsigned int wave = threadIdx.x >> 5;
  const unsigned int tile = blockIdx.x * 8u + wave;      // < 256*8 = 2048
  const unsigned int mt = tile >> 3, nt = tile & 7u;
  const unsigned int row0 = mt * 32u, col0 = nt * 64u;
  const unsigned int l15 = lane & 15u;
  const unsigned int halfsel = lane >> 4;
  const unsigned int kA = halfsel * 8u;
  const unsigned int kB = halfsel * 8u;

  v8f acc[2][4];
#pragma unroll
  for (unsigned int mi = 0; mi < 2u; ++mi)
#pragma unroll
    for (unsigned int t = 0; t < 4u; ++t)
      acc[mi][t] = (v8f){0.f, 0.f, 0.f, 0.f, 0.f, 0.f, 0.f, 0.f};

  const unsigned short* apr0 = Aa + (row0 + l15) * 512u + kA;
  const unsigned short* apr1 = apr0 + 16u * 512u;
  const unsigned short* bpr  = Bb + (col0 + l15) * 512u + kB;

  for (unsigned int k0 = 0; k0 < 512u; k0 += 32u) {
    ABfrag a0, a1;
    a0.q[0] = *(const uint4*)(apr0 + k0);
    a0.q[1] = *(const uint4*)(apr0 + k0 + 16);
    a1.q[0] = *(const uint4*)(apr1 + k0);
    a1.q[1] = *(const uint4*)(apr1 + k0 + 16);
#pragma unroll
    for (unsigned int t = 0; t < 4u; ++t) {
      ABfrag b;
      const unsigned short* bp = bpr + t * (16u * 512u) + k0;
      b.q[0] = *(const uint4*)(bp);
      b.q[1] = *(const uint4*)(bp + 16);
      acc[0][t] = wmma_bf16(a0.v, b.v, acc[0][t]);
      acc[1][t] = wmma_bf16(a1.v, b.v, acc[1][t]);
    }
  }

#pragma unroll
  for (unsigned int t = 0; t < 4u; ++t) {
    const unsigned int ncol = col0 + t * 16u + l15;
    const float bo = b_out[ncol];
#pragma unroll
    for (unsigned int mi = 0; mi < 2u; ++mi) {
#pragma unroll
      for (unsigned int r = 0; r < 8u; ++r) {
        const unsigned int M = mi * 16u + r + (halfsel << 3);
        out[(row0 + M) * 512u + ncol] = acc[mi][t][r] + bo;
      }
    }
  }
}

// ---------------------------------------------------------------------------
extern "C" void kernel_launch(void* const* d_in, const int* in_sizes, int n_in,
                              void* d_out, int out_size, void* d_ws,
                              size_t ws_size, hipStream_t stream) {
  (void)in_sizes; (void)n_in; (void)out_size; (void)ws_size;
  const float* x     = (const float*)d_in[0];
  const float* w_qkv = (const float*)d_in[1];
  const float* w_out = (const float*)d_in[2];
  const float* b_out = (const float*)d_in[3];
  const float* rpe   = (const float*)d_in[4];
  const int*   hop   = (const int*)d_in[5];
  float* out = (float*)d_out;
  unsigned short* W = (unsigned short*)d_ws;

  k_convert<<<2048, 256, 0, stream>>>(x, w_qkv, w_out, W);
  k_qkv<<<768, 256, 0, stream>>>(W);
  k_attn<<<512, 32, LDS_TOTAL, stream>>>(rpe, hop, W, out);
  k_out<<<256, 256, 0, stream>>>(W, b_out, out);
}
